// Attention_85444079386748
// MI455X (gfx1250) — compile-verified, matrix operand order checked by reading, not run on hardware
//
#include <hip/hip_runtime.h>
#include <hip/hip_bf16.h>
#include <math.h>

// ---------------------------------------------------------------------------
// MHA forward  B=2,N=4096,QDIM=768,H=8,D=64  for gfx1250 (MI455X).
//  - bf16 WMMA (v_wmma_f32_16x16x32_bf16), fp32 accumulate
//  - flash-attention streaming softmax (S never touches HBM)
//  - ALL global->LDS tile movement via GLOBAL_LOAD_ASYNC_TO_LDS_B128 with
//    double buffering (weights pre-transposed; V produced transposed so no
//    per-tile transposes remain in the hot loops)
//  - LDS-staged, fully coalesced b128 epilogue stores
// ---------------------------------------------------------------------------

typedef __bf16 bf16;
typedef __attribute__((ext_vector_type(8)))  __bf16 v8bf;
typedef __attribute__((ext_vector_type(16))) __bf16 v16bf;
typedef __attribute__((ext_vector_type(8)))  float  v8f;
typedef __attribute__((ext_vector_type(4)))  int    v4i;

#define Bdim   2
#define Nseq   4096
#define QDIM   768
#define Hn     8
#define Dh     64
#define INNER  512
#define Mrows  (Bdim * Nseq)   // 8192

#define AS1 __attribute__((address_space(1)))
#define AS3 __attribute__((address_space(3)))

#if defined(__has_builtin)
#if __has_builtin(__builtin_amdgcn_global_load_async_to_lds_b128)
#define HAVE_ASYNC_LDS 1
#endif
#endif

// 16B global -> LDS copy: async (ASYNCcnt) when available, sync fallback.
__device__ inline void copy16(const bf16* g, bf16* l) {
#ifdef HAVE_ASYNC_LDS
    __builtin_amdgcn_global_load_async_to_lds_b128(
        (AS1 v4i*)(void*)g, (AS3 v4i*)(void*)l, 0, 0);
#else
    *(v8bf*)l = *(const v8bf*)g;
#endif
}
__device__ inline void wait_async() {
#ifdef HAVE_ASYNC_LDS
    asm volatile("s_wait_asynccnt 0" ::: "memory");
#endif
}
__device__ inline void wait_lds() {
    asm volatile("s_wait_dscnt 0" ::: "memory");
}

// --- A/B fragment loader (ISA 7.12.2 16-bit A 16x32 layout) ------------------
// lane<16 holds K=0..7,16..23 of its row; lane>=16 holds K=8..15,24..31.
__device__ inline v16bf load_frag(const bf16* base, int koff) {
    v8bf lo = *(const v8bf*)(base + koff);
    v8bf hi = *(const v8bf*)(base + koff + 16);
    v16bf r;
#pragma unroll
    for (int i = 0; i < 8; ++i) { r[i] = lo[i]; r[8 + i] = hi[i]; }
    return r;
}

// --- fp32 -> bf16 conversion, 8 elements/thread -----------------------------
__global__ __launch_bounds__(256) void cvt_f32_bf16(const float* __restrict__ in,
                                                    bf16* __restrict__ out, int n8) {
    int i = blockIdx.x * 256 + threadIdx.x;
    if (i < n8) {
        v8f v = *(const v8f*)(in + (size_t)i * 8);
        *(v8bf*)(out + (size_t)i * 8) = __builtin_convertvector(v, v8bf);
    }
}

// --- fp32 -> bf16 transposing conversion: out[n][k] = in[k][n] ---------------
// grid = (K/256, Nc); coalesced writes along k.
__global__ __launch_bounds__(256) void cvtT_f32_bf16(const float* __restrict__ in,
                                                     bf16* __restrict__ out,
                                                     int K, int Nc) {
    int k = blockIdx.x * 256 + threadIdx.x;
    int n = blockIdx.y;
    if (k < K) out[(size_t)n * K + k] = (bf16)in[(size_t)k * Nc + n];
}

// --- bf16 GEMM: Y[M x Nc] = Xa[M x K] * Xb[Nc x K]^T -------------------------
// Both inputs K-major, so A and B tiles are identical contiguous async copies.
// outMode 0: bf16 scatter to [B,H,N,D] (Q/K proj, *scale), LDS-staged coalesced
// outMode 1: fp32 row-major + bias (final projection)
// outMode 2: bf16 row-major (V^T production), LDS-staged coalesced
#define BM 128
#define BN 128
#define BK 32
#define LDT 40   // padded LDS row stride for 32-wide tiles (80B, 16B aligned)
#define LDC 72   // padded stride for 64-wide epilogue tile (144B, 16B aligned)

union SharedGemm {
    struct { bf16 As[2][BM][LDT]; bf16 Bt[2][BN][LDT]; } t;  // 40960 B
    bf16 Cs[8][32][LDC];                                     // 36864 B
};

__global__ __launch_bounds__(256)
void gemm_bf16(const bf16* __restrict__ Xa, const bf16* __restrict__ Xb,
               int M, int K, int Nc, float scale, int outMode,
               bf16* __restrict__ out_bf, float* __restrict__ out_f32,
               const float* __restrict__ bias) {
    __shared__ SharedGemm sh;

    const int tid  = threadIdx.x;
    const int lane = tid & 31;
    const int w    = tid >> 5;     // 0..7
    const int wm   = w >> 1;       // 0..3 -> 32 rows each
    const int wn   = w & 1;        // 0..1 -> 64 cols each
    const int hf   = lane >> 4;
    const int l15  = lane & 15;
    const int koff = hf * 8;
    const int m0   = blockIdx.x * BM;
    const int n0   = blockIdx.y * BN;

    const int row = tid >> 1, seg = tid & 1;
    auto loadA = [&](int k0, int nb) {
        const bf16* src = Xa + (size_t)(m0 + row) * K + k0 + seg * 16;
        copy16(src,     &sh.t.As[nb][row][seg * 16]);
        copy16(src + 8, &sh.t.As[nb][row][seg * 16 + 8]);
    };
    auto loadB = [&](int k0, int nb) {
        const bf16* src = Xb + (size_t)(n0 + row) * K + k0 + seg * 16;
        copy16(src,     &sh.t.Bt[nb][row][seg * 16]);
        copy16(src + 8, &sh.t.Bt[nb][row][seg * 16 + 8]);
    };

    v8f acc[2][4] = {};

    loadA(0, 0); loadB(0, 0);
    wait_async();
    __syncthreads();

    int buf = 0;
    for (int k0 = 0; k0 < K; k0 += BK) {
        if (k0 + BK < K) { loadA(k0 + BK, buf ^ 1); loadB(k0 + BK, buf ^ 1); }

        v16bf afr[2], bfr[4];
#pragma unroll
        for (int mt = 0; mt < 2; ++mt)
            afr[mt] = load_frag(&sh.t.As[buf][wm * 32 + mt * 16 + l15][0], koff);
#pragma unroll
        for (int nt = 0; nt < 4; ++nt)
            bfr[nt] = load_frag(&sh.t.Bt[buf][wn * 64 + nt * 16 + l15][0], koff);
#pragma unroll
        for (int mt = 0; mt < 2; ++mt)
#pragma unroll
            for (int nt = 0; nt < 4; ++nt)
                acc[mt][nt] = __builtin_amdgcn_wmma_f32_16x16x32_bf16(
                    false, afr[mt], false, bfr[nt], (short)0, acc[mt][nt], false, false);

        wait_async();        // prefetch for next tile has landed
        __syncthreads();
        buf ^= 1;
    }

    if (outMode != 1) {
        // stage wave's 32x64 slab in LDS (C layout: VGPR r -> M=r+8*hf, N=l15)
#pragma unroll
        for (int mt = 0; mt < 2; ++mt)
#pragma unroll
            for (int nt = 0; nt < 4; ++nt)
#pragma unroll
                for (int r = 0; r < 8; ++r)
                    sh.Cs[w][mt * 16 + r + 8 * hf][nt * 16 + l15] =
                        (bf16)(acc[mt][nt][r] * scale);
        wait_lds();          // intra-wave LDS RAW only
        int gm = m0 + wm * 32 + lane;
        bf16* dst;
        if (outMode == 0) {
            // wave's 64 columns are exactly one head -> contiguous in [B,H,N,D]
            int b = gm >> 12, n = gm & (Nseq - 1);
            int h = (n0 + wn * 64) >> 6;
            dst = out_bf + (((size_t)(b * Hn + h) * Nseq) + n) * Dh;
        } else {
            dst = out_bf + (size_t)gm * Nc + n0 + wn * 64;
        }
#pragma unroll
        for (int c = 0; c < 8; ++c)
            *(v8bf*)(dst + c * 8) = *(const v8bf*)&sh.Cs[w][lane][c * 8];
    } else {
#pragma unroll
        for (int nt = 0; nt < 4; ++nt) {
            int gc = n0 + wn * 64 + nt * 16 + l15;
            float bv = bias[gc];
#pragma unroll
            for (int mt = 0; mt < 2; ++mt)
#pragma unroll
                for (int r = 0; r < 8; ++r) {
                    int gm = m0 + wm * 32 + mt * 16 + r + 8 * hf;
                    out_f32[(size_t)gm * Nc + gc] = acc[mt][nt][r] + bv;
                }
        }
    }
}

// --- flash attention ---------------------------------------------------------
// grid = (N/128, B*H); block = 256 (8 waves). Wave w owns S/O rows 16w..16w+15.
// K is [B,H,N,D] (row-major K tile == N-major B^T for S = Q*K^T).
// V is pre-transposed globally: vT[H*D, B*N], so V tiles are plain copies.
#define ALD 80   // padded row stride for 64-wide tiles (160B, 16B aligned)

__global__ __launch_bounds__(256)
void flash_attn(const bf16* __restrict__ Qg, const bf16* __restrict__ Kg,
                const bf16* __restrict__ VTg, bf16* __restrict__ Og) {
    __shared__ bf16 QP[128][ALD];       // Q tile, then reused for P
    __shared__ bf16 Ks[2][64][ALD];     // double buffered
    __shared__ bf16 Vt[2][64][ALD];     // Vt[d][j]

    const int tid  = threadIdx.x;
    const int lane = tid & 31;
    const int w    = tid >> 5;
    const int hf   = lane >> 4;
    const int l15  = lane & 15;
    const int koff = hf * 8;
    const int qb   = blockIdx.x;
    const int bh   = blockIdx.y;
    const int b    = bh >> 3, h = bh & 7;
    const size_t base = (size_t)bh * Nseq * Dh;

    // Q tile 128x64: each wave loads exactly its own 16 rows.
    {
        int qrow = tid >> 1, qseg = tid & 1;
        const bf16* src = Qg + base + (size_t)(qb * 128 + qrow) * Dh + qseg * 32;
#pragma unroll
        for (int c = 0; c < 4; ++c)
            copy16(src + c * 8, &QP[qrow][qseg * 32 + c * 8]);
    }
    wait_async();
    wait_lds();

    v16bf qfr[2];
    qfr[0] = load_frag(&QP[w * 16 + l15][0], koff);
    qfr[1] = load_frag(&QP[w * 16 + l15][0], 32 + koff);

    const int row = tid >> 2, seg = tid & 3;   // 64 rows x 4 16-elem segments
    auto loadKV = [&](int j0, int nb) {
        const bf16* ks = Kg + base + (size_t)(j0 + row) * Dh + seg * 16;
        copy16(ks,     &Ks[nb][row][seg * 16]);
        copy16(ks + 8, &Ks[nb][row][seg * 16 + 8]);
        // vT row = h*Dh + d, col = b*Nseq + j
        const bf16* vs = VTg + (size_t)(h * Dh + row) * Mrows + b * Nseq + j0 + seg * 16;
        copy16(vs,     &Vt[nb][row][seg * 16]);
        copy16(vs + 8, &Vt[nb][row][seg * 16 + 8]);
    };

    float mst[8], lst[8];
#pragma unroll
    for (int r = 0; r < 8; ++r) { mst[r] = -1e30f; lst[r] = 0.f; }
    v8f oacc[4] = {};

    loadKV(0, 0);
    wait_async();
    __syncthreads();

    int buf = 0;
    for (int j0 = 0; j0 < Nseq; j0 += 64) {
        if (j0 + 64 < Nseq) loadKV(j0 + 64, buf ^ 1);   // prefetch next tile

        // S = Q * K^T (scale folded into Q at projection time)
        v8f s[4];
#pragma unroll
        for (int nt = 0; nt < 4; ++nt) {
            v16bf b0 = load_frag(&Ks[buf][nt * 16 + l15][0], koff);
            v16bf b1 = load_frag(&Ks[buf][nt * 16 + l15][0], 32 + koff);
            v8f z = {};
            z = __builtin_amdgcn_wmma_f32_16x16x32_bf16(false, qfr[0], false, b0,
                                                        (short)0, z, false, false);
            s[nt] = __builtin_amdgcn_wmma_f32_16x16x32_bf16(false, qfr[1], false, b1,
                                                            (short)0, z, false, false);
        }

        // online softmax; a row spans 16 lanes of each half-wave
        float rmax[8];
#pragma unroll
        for (int r = 0; r < 8; ++r)
            rmax[r] = fmaxf(fmaxf(s[0][r], s[1][r]), fmaxf(s[2][r], s[3][r]));
#pragma unroll
        for (int off = 1; off < 16; off <<= 1)
#pragma unroll
            for (int r = 0; r < 8; ++r)
                rmax[r] = fmaxf(rmax[r], __shfl_xor(rmax[r], off, 16));

        float fac[8], rsum[8];
#pragma unroll
        for (int r = 0; r < 8; ++r) {
            float mn = fmaxf(mst[r], rmax[r]);
            fac[r] = __expf(mst[r] - mn);
            mst[r] = mn;
            rsum[r] = 0.f;
        }
#pragma unroll
        for (int nt = 0; nt < 4; ++nt)
#pragma unroll
            for (int r = 0; r < 8; ++r) {
                float p = __expf(s[nt][r] - mst[r]);
                s[nt][r] = p;
                rsum[r] += p;
            }
#pragma unroll
        for (int off = 1; off < 16; off <<= 1)
#pragma unroll
            for (int r = 0; r < 8; ++r)
                rsum[r] += __shfl_xor(rsum[r], off, 16);
#pragma unroll
        for (int r = 0; r < 8; ++r) lst[r] = lst[r] * fac[r] + rsum[r];
#pragma unroll
        for (int dt = 0; dt < 4; ++dt)
#pragma unroll
            for (int r = 0; r < 8; ++r) oacc[dt][r] *= fac[r];

        // P: C-layout regs -> A-layout LDS (wave-private rows of QP)
#pragma unroll
        for (int nt = 0; nt < 4; ++nt)
#pragma unroll
            for (int r = 0; r < 8; ++r)
                QP[w * 16 + r + 8 * hf][nt * 16 + l15] = (bf16)s[nt][r];
        wait_lds();   // intra-wave LDS RAW

        // O += P * V
#pragma unroll
        for (int dt = 0; dt < 4; ++dt)
#pragma unroll
            for (int ks2 = 0; ks2 < 2; ++ks2) {
                v16bf pa = load_frag(&QP[w * 16 + l15][0], ks2 * 32 + koff);
                v16bf vb = load_frag(&Vt[buf][dt * 16 + l15][0], ks2 * 32 + koff);
                oacc[dt] = __builtin_amdgcn_wmma_f32_16x16x32_bf16(
                    false, pa, false, vb, (short)0, oacc[dt], false, false);
            }

        wait_async();      // next K/V tile has landed
        __syncthreads();
        buf ^= 1;
    }

    // normalize, write O as [B*N, INNER] bf16 (head-major columns)
#pragma unroll
    for (int dt = 0; dt < 4; ++dt)
#pragma unroll
        for (int r = 0; r < 8; ++r) {
            float inv = 1.0f / lst[r];
            int n = qb * 128 + w * 16 + r + 8 * hf;
            int c = h * Dh + dt * 16 + l15;
            Og[((size_t)(b * Nseq + n)) * INNER + c] = (bf16)(oacc[dt][r] * inv);
        }
}

// ---------------------------------------------------------------------------
extern "C" void kernel_launch(void* const* d_in, const int* in_sizes, int n_in,
                              void* d_out, int out_size, void* d_ws, size_t ws_size,
                              hipStream_t stream) {
    const float* x    = (const float*)d_in[0];
    const float* Wq   = (const float*)d_in[1];
    const float* Wk   = (const float*)d_in[2];
    const float* Wv   = (const float*)d_in[3];
    const float* Wout = (const float*)d_in[4];
    const float* bo   = (const float*)d_in[5];

    char* ws = (char*)d_ws;
    size_t off = 0;
    bf16* x_bf  = (bf16*)(ws + off); off += (size_t)Mrows * QDIM * 2;
    bf16* wqT   = (bf16*)(ws + off); off += (size_t)QDIM * INNER * 2;  // [INNER][QDIM]
    bf16* wkT   = (bf16*)(ws + off); off += (size_t)QDIM * INNER * 2;
    bf16* wvT   = (bf16*)(ws + off); off += (size_t)QDIM * INNER * 2;
    bf16* woT   = (bf16*)(ws + off); off += (size_t)INNER * QDIM * 2;  // [QDIM][INNER]
    bf16* q_bf  = (bf16*)(ws + off); off += (size_t)Mrows * INNER * 2; // [B,H,N,D]
    bf16* k_bf  = (bf16*)(ws + off); off += (size_t)Mrows * INNER * 2; // [B,H,N,D]
    bf16* vT_bf = (bf16*)(ws + off); off += (size_t)Mrows * INNER * 2; // [H*D, B*N]
    bf16* o_bf  = (bf16*)(ws + off); off += (size_t)Mrows * INNER * 2; // [B*N, INNER]
    (void)off; (void)ws_size; (void)in_sizes; (void)n_in; (void)out_size;

    dim3 blk(256);
    const int nx8 = Mrows * QDIM / 8;
    cvt_f32_bf16<<<(nx8 + 255) / 256, blk, 0, stream>>>(x, x_bf, nx8);
    // weights: transpose to K-major during conversion
    cvtT_f32_bf16<<<dim3(QDIM / 256, INNER), blk, 0, stream>>>(Wq, wqT, QDIM, INNER);
    cvtT_f32_bf16<<<dim3(QDIM / 256, INNER), blk, 0, stream>>>(Wk, wkT, QDIM, INNER);
    cvtT_f32_bf16<<<dim3(QDIM / 256, INNER), blk, 0, stream>>>(Wv, wvT, QDIM, INNER);
    cvtT_f32_bf16<<<dim3(INNER / 256, QDIM), blk, 0, stream>>>(Wout, woT, INNER, QDIM);

    const float scale = 0.125f;  // 1/sqrt(64), folded into Q
    dim3 gqkv(Mrows / BM, INNER / BN);
    gemm_bf16<<<gqkv, blk, 0, stream>>>(x_bf, wqT, Mrows, QDIM, INNER, scale, 0,
                                        q_bf, nullptr, nullptr);
    gemm_bf16<<<gqkv, blk, 0, stream>>>(x_bf, wkT, Mrows, QDIM, INNER, 1.0f, 0,
                                        k_bf, nullptr, nullptr);
    // V^T = (x Wv)^T : A = Wv^T [INNER x QDIM], B-tiles straight from x_bf
    dim3 gvt(INNER / BM, Mrows / BN);
    gemm_bf16<<<gvt, blk, 0, stream>>>(wvT, x_bf, INNER, QDIM, Mrows, 1.0f, 2,
                                       vT_bf, nullptr, nullptr);

    dim3 gatt(Nseq / 128, Bdim * Hn);
    flash_attn<<<gatt, blk, 0, stream>>>(q_bf, k_bf, vT_bf, o_bf);

    dim3 gout(Mrows / BM, QDIM / BN);
    gemm_bf16<<<gout, blk, 0, stream>>>(o_bf, woT, Mrows, INNER, QDIM, 1.0f, 1,
                                        nullptr, (float*)d_out, bo);
}
